// MaskedMultiHeadAttentionWithLoRAFinetuning_17566416240822
// MI455X (gfx1250) — matrix operand order repaired and checked by reading, not compiled, not run on hardware
//
#include <hip/hip_runtime.h>
#include <hip/hip_bf16.h>

typedef __attribute__((ext_vector_type(16))) __bf16 v16bf;
typedef __attribute__((ext_vector_type(8)))  __bf16 v8bf;
typedef __attribute__((ext_vector_type(8)))  float  v8f;

#define BB 4
#define SS 2048
#define DMODEL 1024
#define HH 16
#define DD 64
#define HD 1024
#define MM (BB*SS)

// ---------- WMMA helpers (CDNA5 16x16x32 bf16, f32 accum) ----------

__device__ __forceinline__ v8f wmma_bf16(v16bf a, v16bf b, v8f c) {
  return __builtin_amdgcn_wmma_f32_16x16x32_bf16(false, a, false, b, (short)0, c, false, false);
}

__device__ __forceinline__ v16bf cat8(v8bf lo, v8bf hi) {
  return __builtin_shufflevector(lo, hi, 0,1,2,3,4,5,6,7,8,9,10,11,12,13,14,15);
}

// A fragment: 16x32 bf16, row-major source. lane<16 -> row=lane, K in {grp*8.., 16+grp*8..}
__device__ __forceinline__ v16bf load_a_frag(const __bf16* __restrict__ A, int lda,
                                             int row0, int k0, int lane) {
  const int grp = lane >> 4, m = lane & 15;
  const __bf16* p = A + (size_t)(row0 + m) * lda + k0 + grp * 8;
  return cat8(*(const v8bf*)p, *(const v8bf*)(p + 16));
}

// B fragment: 32x16 bf16 from [N,K]-stored operand (column n contiguous over K).
__device__ __forceinline__ v16bf load_b_frag(const __bf16* __restrict__ Bt, int ldb,
                                             int col0, int k0, int lane) {
  const int n = lane & 15, kh = (lane >> 4) * 8;
  const __bf16* p = Bt + (size_t)(col0 + n) * ldb + k0 + kh;
  return cat8(*(const v8bf*)p, *(const v8bf*)(p + 16));
}

// ---------- CDNA5 async global->LDS copy (ASYNCcnt) ----------
__device__ __forceinline__ void async_g2l_b128(void* lds_ptr, unsigned long long gbase,
                                               unsigned int voff) {
  unsigned int lds = (unsigned int)(size_t)lds_ptr;
  asm volatile("global_load_async_to_lds_b128 %0, %1, %2"
               :: "v"(lds), "v"(voff), "s"(gbase) : "memory");
}
__device__ __forceinline__ void wait_async0() {
  asm volatile("s_wait_asynccnt 0x0" ::: "memory");
}

// ---------- prep kernels ----------

__global__ __launch_bounds__(256) void conv_bf16(const float* __restrict__ x,
                                                 __bf16* __restrict__ y, int n) {
  int i = blockIdx.x * blockDim.x + threadIdx.x;
  if (i < n) y[i] = (__bf16)x[i];
}

__global__ __launch_bounds__(256) void transpose_bf16(const float* __restrict__ W,
                                                      __bf16* __restrict__ Wt, int K, int N) {
  __shared__ float tile[32][33];
  const int k0 = blockIdx.y * 32, n0 = blockIdx.x * 32;
  const int tx = threadIdx.x, ty = threadIdx.y;
  #pragma unroll
  for (int j = 0; j < 32; j += 8)
    tile[ty + j][tx] = W[(size_t)(k0 + ty + j) * N + n0 + tx];
  __syncthreads();
  #pragma unroll
  for (int j = 0; j < 32; j += 8)
    Wt[(size_t)(n0 + ty + j) * K + k0 + tx] = (__bf16)tile[tx][ty + j];
}

__global__ __launch_bounds__(256) void conv_loraB(const float* __restrict__ Bw,
                                                  __bf16* __restrict__ Bt, int N) {
  int n = blockIdx.x * blockDim.x + threadIdx.x;
  if (n >= N) return;
  #pragma unroll
  for (int r = 0; r < 8; ++r)  Bt[(size_t)n * 32 + r] = (__bf16)Bw[(size_t)r * N + n];
  #pragma unroll
  for (int r = 8; r < 32; ++r) Bt[(size_t)n * 32 + r] = (__bf16)0.0f;
}

__global__ __launch_bounds__(256) void lora_rankA(const float* __restrict__ X,
                                                  const float* __restrict__ Aw,
                                                  const float* __restrict__ Ab,
                                                  __bf16* __restrict__ la, int M, int K) {
  const int lane = threadIdx.x & 31;
  const int row  = blockIdx.x * 8 + (threadIdx.x >> 5);
  if (row >= M) return;
  const float* x = X + (size_t)row * K;
  float acc[8] = {0,0,0,0,0,0,0,0};
  for (int kk = lane; kk < K; kk += 32) {
    const float xv = x[kk];
    const float* ar = Aw + (size_t)kk * 8;
    #pragma unroll
    for (int r = 0; r < 8; ++r) acc[r] += xv * ar[r];
  }
  #pragma unroll
  for (int r = 0; r < 8; ++r) {
    #pragma unroll
    for (int off = 16; off > 0; off >>= 1)
      acc[r] += __shfl_xor(acc[r], off, 32);
  }
  if (lane == 0) {
    #pragma unroll
    for (int r = 0; r < 8; ++r)  la[(size_t)row * 32 + r] = (__bf16)(acc[r] + Ab[r]);
    #pragma unroll
    for (int r = 8; r < 32; ++r) la[(size_t)row * 32 + r] = (__bf16)0.0f;
  }
}

// ---------- main GEMM: async-LDS double-buffered, 64x128 block tile ----------
template<int MODE, bool LORA>
__global__ __launch_bounds__(256) void gemm_bf16_wmma(
    const __bf16* __restrict__ A, const __bf16* __restrict__ Bt,
    const __bf16* __restrict__ laA, const __bf16* __restrict__ laBt,
    const float* __restrict__ bias1, const float* __restrict__ bias2,
    void* __restrict__ outp, int M, int N, int K) {
  __shared__ __attribute__((aligned(16))) __bf16 Abuf[2][64][40];
  __shared__ __attribute__((aligned(16))) __bf16 Bbuf[2][128][40];
  const int tid  = threadIdx.x;
  const int lane = tid & 31;
  const int wave = tid >> 5;
  const int wr = wave >> 1, wc = wave & 1;
  const int rowblk = blockIdx.y * 64;
  const int colblk = blockIdx.x * 128;

  const int arow = tid >> 2, aseg = tid & 3;
  const unsigned int avoff = (unsigned int)(((rowblk + arow) * K + aseg * 8) * 2);
  const int bn0 = tid >> 2,          bs0 = tid & 3;
  const int bn1 = (tid + 256) >> 2,  bs1 = tid & 3;
  const unsigned int bvoff0 = (unsigned int)(((colblk + bn0) * K + bs0 * 8) * 2);
  const unsigned int bvoff1 = (unsigned int)(((colblk + bn1) * K + bs1 * 8) * 2);

  auto issue = [&](int buf, int k0) {
    const unsigned long long ab = (unsigned long long)(size_t)A  + (unsigned long long)(k0 * 2);
    const unsigned long long bb = (unsigned long long)(size_t)Bt + (unsigned long long)(k0 * 2);
    async_g2l_b128(&Abuf[buf][arow][aseg * 8], ab, avoff);
    async_g2l_b128(&Bbuf[buf][bn0][bs0 * 8],  bb, bvoff0);
    async_g2l_b128(&Bbuf[buf][bn1][bs1 * 8],  bb, bvoff1);
  };

  v8f acc[4] = {v8f{}, v8f{}, v8f{}, v8f{}};
  const int m = lane & 15, grp = lane >> 4;

  issue(0, 0);
  for (int k0 = 0; k0 < K; k0 += 32) {
    const int buf = (k0 >> 5) & 1;
    wait_async0();
    __syncthreads();
    if (k0 + 32 < K) issue(buf ^ 1, k0 + 32);
    const __bf16* ar = &Abuf[buf][wr * 16 + m][grp * 8];
    v16bf af = cat8(*(const v8bf*)ar, *(const v8bf*)(ar + 16));
    #pragma unroll
    for (int t = 0; t < 4; ++t) {
      const __bf16* br = &Bbuf[buf][wc * 64 + t * 16 + m][grp * 8];
      v16bf bfr = cat8(*(const v8bf*)br, *(const v8bf*)(br + 16));
      acc[t] = wmma_bf16(af, bfr, acc[t]);
    }
    __syncthreads();
  }

  if (LORA) {
    v16bf af = load_a_frag(laA, 32, rowblk + wr * 16, 0, lane);
    #pragma unroll
    for (int t = 0; t < 4; ++t) {
      v16bf bfr = load_b_frag(laBt, 32, colblk + wc * 64 + t * 16, 0, lane);
      acc[t] = wmma_bf16(af, bfr, acc[t]);
    }
  }

  const int mbase = rowblk + wr * 16 + 8 * grp;
  #pragma unroll
  for (int t = 0; t < 4; ++t) {
    const int n = colblk + wc * 64 + t * 16 + m;
    float bsum = 0.f;
    if (bias1) bsum += bias1[n];
    if (bias2) bsum += bias2[n];
    if (MODE == 2) {
      float* O = (float*)outp;
      #pragma unroll
      for (int i = 0; i < 8; ++i) O[(size_t)(mbase + i) * N + n] = acc[t][i] + bsum;
    } else {
      __bf16* O = (__bf16*)outp;
      const int h = n >> 6, d = n & 63;
      #pragma unroll
      for (int i = 0; i < 8; ++i) {
        const int mi = mbase + i, b = mi >> 11, s = mi & (SS - 1);
        const float v = acc[t][i] + bsum;
        if (MODE == 0) O[(size_t)((b * HH + h) * SS + s) * DD + d] = (__bf16)v;
        else           O[(size_t)((b * HH + h) * DD + d) * SS + s] = (__bf16)v;
      }
    }
  }
}

// ---------- pass 1: per-COLUMN (query-axis) softmax stats over causal region ----------
// Diagonal tile peeled (masked); all remaining q-tiles strictly below diagonal -> unmasked.
__global__ __launch_bounds__(256) void attn_col_stats(
    const __bf16* __restrict__ qT, const __bf16* __restrict__ kT,
    float* __restrict__ colmax, float* __restrict__ colinv) {
  const int lane = threadIdx.x & 31;
  const int wave = threadIdx.x >> 5;
  const int bh = blockIdx.y;
  const int k0 = (blockIdx.x * 8 + wave) * 16;
  const __bf16* qb = qT + (size_t)bh * SS * DD;
  const __bf16* kb = kT + (size_t)bh * SS * DD;
  const v16bf bk0 = load_b_frag(kb, DD, k0, 0,  lane);
  const v16bf bk1 = load_b_frag(kb, DD, k0, 32, lane);
  const int nloc = lane & 15, rgrp = (lane >> 4) * 8;
  float runm = -3.0e38f, runs = 0.f;

  { // diagonal tile q0 == k0 (only tile that needs the causal mask)
    v16bf a0 = load_a_frag(qb, DD, k0, 0,  lane);
    v16bf a1 = load_a_frag(qb, DD, k0, 32, lane);
    v8f c = {};
    c = wmma_bf16(a0, bk0, c);
    c = wmma_bf16(a1, bk1, c);
    float tm = -3.0e38f;
    #pragma unroll
    for (int i = 0; i < 8; ++i)
      tm = fmaxf(tm, (rgrp + i >= nloc) ? c[i] : -3.0e38f);
    runm = tm;
    #pragma unroll
    for (int i = 0; i < 8; ++i)
      runs += (rgrp + i >= nloc) ? __expf(c[i] - runm) : 0.f;
  }

  for (int q0 = k0 + 16; q0 < SS; q0 += 16) {  // strictly-below tiles: unmasked
    v16bf a0 = load_a_frag(qb, DD, q0, 0,  lane);
    v16bf a1 = load_a_frag(qb, DD, q0, 32, lane);
    v8f c = {};
    c = wmma_bf16(a0, bk0, c);
    c = wmma_bf16(a1, bk1, c);
    float tm = c[0];
    #pragma unroll
    for (int i = 1; i < 8; ++i) tm = fmaxf(tm, c[i]);
    const float nm = fmaxf(runm, tm);
    runs *= __expf(runm - nm);
    runm = nm;
    #pragma unroll
    for (int i = 0; i < 8; ++i) runs += __expf(c[i] - nm);
  }

  const float om = __shfl_xor(runm, 16, 32);
  const float os = __shfl_xor(runs, 16, 32);
  const float nm = fmaxf(runm, om);
  const float s  = runs * __expf(runm - nm) + os * __expf(om - nm);
  if (lane < 16) {
    colmax[(size_t)bh * SS + k0 + lane] = nm;
    colinv[(size_t)bh * SS + k0 + lane] = 0.125f / s;   // fold post-softmax 1/sqrt(D)
  }
}

// ---------- pass 2: recompute scores, normalize per-column, P@V ----------
// Fully-valid k-tiles (kc < q0) run unmasked; only the tail 32-key step masks.
__global__ __launch_bounds__(256) void attn_apply(
    const __bf16* __restrict__ qT, const __bf16* __restrict__ kT,
    const __bf16* __restrict__ vT,
    const float* __restrict__ colmax, const float* __restrict__ colinv,
    __bf16* __restrict__ attn_out) {
  __shared__ __attribute__((aligned(32))) __bf16 pbuf[8][16][40];
  const int lane = threadIdx.x & 31;
  const int wave = threadIdx.x >> 5;
  const int bh = blockIdx.y * 8 + wave;
  const int q0 = blockIdx.x * 16;
  const __bf16* qb = qT + (size_t)bh * SS * DD;
  const __bf16* kb = kT + (size_t)bh * SS * DD;
  const __bf16* vb = vT + (size_t)bh * DD * SS;
  const float* cm = colmax + (size_t)bh * SS;
  const float* ci = colinv + (size_t)bh * SS;
  const v16bf aq0 = load_a_frag(qb, DD, q0, 0,  lane);
  const v16bf aq1 = load_a_frag(qb, DD, q0, 32, lane);
  v8f o0 = {}, o1 = {}, o2 = {}, o3 = {};
  const int nloc = lane & 15, rgrp = (lane >> 4) * 8, grp = lane >> 4;

  const int kend = (q0 & 31) ? (q0 - 16) : q0;   // start of the (masked) tail step

  for (int k0 = 0; k0 < kend; k0 += 32) {        // fully-valid region: no masks
    #pragma unroll
    for (int sub = 0; sub < 2; ++sub) {
      const int kc = k0 + sub * 16;
      v16bf b0 = load_b_frag(kb, DD, kc, 0,  lane);
      v16bf b1 = load_b_frag(kb, DD, kc, 32, lane);
      v8f c = {};
      c = wmma_bf16(aq0, b0, c);
      c = wmma_bf16(aq1, b1, c);
      const float cmx = cm[kc + nloc];
      const float civ = ci[kc + nloc];
      #pragma unroll
      for (int i = 0; i < 8; ++i)
        pbuf[wave][rgrp + i][sub * 16 + nloc] = (__bf16)(__expf(c[i] - cmx) * civ);
    }
    __syncthreads();
    const __bf16* prow = &pbuf[wave][nloc][0];
    v16bf pf = cat8(*(const v8bf*)(prow + grp * 8), *(const v8bf*)(prow + 16 + grp * 8));
    o0 = wmma_bf16(pf, load_b_frag(vb, SS, 0,  k0, lane), o0);
    o1 = wmma_bf16(pf, load_b_frag(vb, SS, 16, k0, lane), o1);
    o2 = wmma_bf16(pf, load_b_frag(vb, SS, 32, k0, lane), o2);
    o3 = wmma_bf16(pf, load_b_frag(vb, SS, 48, k0, lane), o3);
    __syncthreads();
  }

  { // tail step: contains the diagonal tile (masked) and possibly an all-zero tile
    const int k0 = kend;
    #pragma unroll
    for (int sub = 0; sub < 2; ++sub) {
      const int kc = k0 + sub * 16;
      if (kc <= q0) {            // uniform branch: EXEC stays all-ones at WMMA
        v16bf b0 = load_b_frag(kb, DD, kc, 0,  lane);
        v16bf b1 = load_b_frag(kb, DD, kc, 32, lane);
        v8f c = {};
        c = wmma_bf16(aq0, b0, c);
        c = wmma_bf16(aq1, b1, c);
        const float cmx = cm[kc + nloc];
        const float civ = ci[kc + nloc];
        #pragma unroll
        for (int i = 0; i < 8; ++i) {
          const float p = (q0 + rgrp + i >= kc + nloc) ? __expf(c[i] - cmx) * civ : 0.f;
          pbuf[wave][rgrp + i][sub * 16 + nloc] = (__bf16)p;
        }
      } else {
        #pragma unroll
        for (int i = 0; i < 8; ++i)
          pbuf[wave][rgrp + i][sub * 16 + nloc] = (__bf16)0.0f;
      }
    }
    __syncthreads();
    const __bf16* prow = &pbuf[wave][nloc][0];
    v16bf pf = cat8(*(const v8bf*)(prow + grp * 8), *(const v8bf*)(prow + 16 + grp * 8));
    o0 = wmma_bf16(pf, load_b_frag(vb, SS, 0,  k0, lane), o0);
    o1 = wmma_bf16(pf, load_b_frag(vb, SS, 16, k0, lane), o1);
    o2 = wmma_bf16(pf, load_b_frag(vb, SS, 32, k0, lane), o2);
    o3 = wmma_bf16(pf, load_b_frag(vb, SS, 48, k0, lane), o3);
  }

  const int b = bh >> 4, h = bh & 15;
  #pragma unroll
  for (int i = 0; i < 8; ++i) {
    const int mi = q0 + rgrp + i;
    const size_t base = ((size_t)(b * SS + mi)) * HD + h * DD + nloc;
    attn_out[base +  0] = (__bf16)o0[i];
    attn_out[base + 16] = (__bf16)o1[i];
    attn_out[base + 32] = (__bf16)o2[i];
    attn_out[base + 48] = (__bf16)o3[i];
  }
}

// ---------- host ----------

extern "C" void kernel_launch(void* const* d_in, const int* in_sizes, int n_in,
                              void* d_out, int out_size, void* d_ws, size_t ws_size,
                              hipStream_t stream) {
  const float* queries = (const float*)d_in[0];
  const float* keys    = (const float*)d_in[1];
  const float* values  = (const float*)d_in[2];
  const float* Wq_w = (const float*)d_in[3];  const float* Wq_b = (const float*)d_in[4];
  const float* Wk_w = (const float*)d_in[5];  const float* Wk_b = (const float*)d_in[6];
  const float* Wv_w = (const float*)d_in[7];  const float* Wv_b = (const float*)d_in[8];
  const float* Aq_w = (const float*)d_in[9];  const float* Aq_b = (const float*)d_in[10];
  const float* Bq_w = (const float*)d_in[11]; const float* Bq_b = (const float*)d_in[12];
  const float* Ak_w = (const float*)d_in[13]; const float* Ak_b = (const float*)d_in[14];
  const float* Bk_w = (const float*)d_in[15]; const float* Bk_b = (const float*)d_in[16];
  const float* Av_w = (const float*)d_in[17]; const float* Av_b = (const float*)d_in[18];
  const float* Bv_w = (const float*)d_in[19]; const float* Bv_b = (const float*)d_in[20];
  const float* Wo_w = (const float*)d_in[21]; const float* Wo_b = (const float*)d_in[22];

  char* wp = (char*)d_ws;
  auto alloc = [&](size_t bytes) -> void* {
    void* p = (void*)wp; wp += (bytes + 255) & ~(size_t)255; return p;
  };
  const size_t XB = (size_t)MM * DMODEL * sizeof(__bf16);
  __bf16* Xq  = (__bf16*)alloc(XB);
  __bf16* Xk  = (__bf16*)alloc(XB);
  __bf16* Xv  = (__bf16*)alloc(XB);
  __bf16* WqT = (__bf16*)alloc((size_t)DMODEL * HD * 2);
  __bf16* WkT = (__bf16*)alloc((size_t)DMODEL * HD * 2);
  __bf16* WvT = (__bf16*)alloc((size_t)DMODEL * HD * 2);
  __bf16* WoT = (__bf16*)alloc((size_t)HD * DMODEL * 2);
  __bf16* BqT = (__bf16*)alloc((size_t)HD * 32 * 2);
  __bf16* BkT = (__bf16*)alloc((size_t)HD * 32 * 2);
  __bf16* BvT = (__bf16*)alloc((size_t)HD * 32 * 2);
  __bf16* laq = (__bf16*)alloc((size_t)MM * 32 * 2);
  __bf16* lak = (__bf16*)alloc((size_t)MM * 32 * 2);
  __bf16* lav = (__bf16*)alloc((size_t)MM * 32 * 2);
  __bf16* qT  = (__bf16*)alloc(XB);                 // [B,H,S,D]
  __bf16* kT  = (__bf16*)alloc(XB);                 // [B,H,S,D]
  __bf16* vT  = (__bf16*)alloc(XB);                 // [B,H,D,S]
  float*  cmx = (float*)alloc((size_t)BB * HH * SS * 4);
  float*  civ = (float*)alloc((size_t)BB * HH * SS * 4);
  __bf16* att = (__bf16*)alloc(XB);                 // [B,S,H*D]

  const int ntot = MM * DMODEL;
  conv_bf16<<<(ntot + 255) / 256, 256, 0, stream>>>(queries, Xq, ntot);
  conv_bf16<<<(ntot + 255) / 256, 256, 0, stream>>>(keys,    Xk, ntot);
  conv_bf16<<<(ntot + 255) / 256, 256, 0, stream>>>(values,  Xv, ntot);

  dim3 tb(32, 8), tg(HD / 32, DMODEL / 32);
  transpose_bf16<<<tg, tb, 0, stream>>>(Wq_w, WqT, DMODEL, HD);
  transpose_bf16<<<tg, tb, 0, stream>>>(Wk_w, WkT, DMODEL, HD);
  transpose_bf16<<<tg, tb, 0, stream>>>(Wv_w, WvT, DMODEL, HD);
  transpose_bf16<<<dim3(DMODEL / 32, HD / 32), tb, 0, stream>>>(Wo_w, WoT, HD, DMODEL);

  conv_loraB<<<HD / 256, 256, 0, stream>>>(Bq_w, BqT, HD);
  conv_loraB<<<HD / 256, 256, 0, stream>>>(Bk_w, BkT, HD);
  conv_loraB<<<HD / 256, 256, 0, stream>>>(Bv_w, BvT, HD);

  lora_rankA<<<MM / 8, 256, 0, stream>>>(queries, Aq_w, Aq_b, laq, MM, DMODEL);
  lora_rankA<<<MM / 8, 256, 0, stream>>>(keys,    Ak_w, Ak_b, lak, MM, DMODEL);
  lora_rankA<<<MM / 8, 256, 0, stream>>>(keys,    Av_w, Av_b, lav, MM, DMODEL); // quirk: Av(keys)

  dim3 gg(HD / 128, MM / 64);
  gemm_bf16_wmma<0, true ><<<gg, 256, 0, stream>>>(Xq, WqT, laq, BqT, Wq_b, Bq_b, qT, MM, HD, DMODEL);
  gemm_bf16_wmma<0, true ><<<gg, 256, 0, stream>>>(Xk, WkT, lak, BkT, Wk_b, Bk_b, kT, MM, HD, DMODEL);
  gemm_bf16_wmma<1, true ><<<gg, 256, 0, stream>>>(Xv, WvT, lav, BvT, Wv_b, Bv_b, vT, MM, HD, DMODEL);

  attn_col_stats<<<dim3(SS / 128, BB * HH), 256, 0, stream>>>(qT, kT, cmx, civ);
  attn_apply<<<dim3(SS / 16, (BB * HH) / 8), 256, 0, stream>>>(qT, kT, vT, cmx, civ, att);

  gemm_bf16_wmma<2, false><<<dim3(DMODEL / 128, MM / 64), 256, 0, stream>>>(
      att, WoT, nullptr, nullptr, Wo_b, nullptr, d_out, MM, DMODEL, HD);
}
